// AiLUT_30829275251111
// MI455X (gfx1250) — matrix operand order repaired and checked by reading, not compiled, not run on hardware
//
#include <hip/hip_runtime.h>
#include <hip/hip_bf16.h>

typedef __attribute__((ext_vector_type(16))) _Float16 v16h;
typedef __attribute__((ext_vector_type(8)))  float    v8f;

#define VV   33
#define VV3  35937   // 33^3

// ---------------------------------------------------------------------------
// Bilinear resize (2048x2048 -> 256x256), matches jnp half-pixel convention.
// ---------------------------------------------------------------------------
__global__ void ail_resize(const float* __restrict__ in, float* __restrict__ out,
                           int BC, int H, int W, int OH, int OW)
{
    int i = blockIdx.x * blockDim.x + threadIdx.x;
    int total = BC * OH * OW;
    if (i >= total) return;
    int ow = i % OW; int t = i / OW;
    int oh = t % OH; int bc = t / OH;

    float sy = (oh + 0.5f) * ((float)H / (float)OH) - 0.5f;
    float sx = (ow + 0.5f) * ((float)W / (float)OW) - 0.5f;
    int y0 = (int)floorf(sy); int x0 = (int)floorf(sx);
    y0 = min(max(y0, 0), H - 1);
    x0 = min(max(x0, 0), W - 1);
    int y1 = min(y0 + 1, H - 1);
    int x1 = min(x0 + 1, W - 1);
    float fy = fminf(fmaxf(sy - (float)y0, 0.f), 1.f);
    float fx = fminf(fmaxf(sx - (float)x0, 0.f), 1.f);

    const float* p = in + (size_t)bc * H * W;
    float r0 = p[(size_t)y0 * W + x0] * (1.f - fx) + p[(size_t)y0 * W + x1] * fx;
    float r1 = p[(size_t)y1 * W + x0] * (1.f - fx) + p[(size_t)y1 * W + x1] * fx;
    out[i] = r0 * (1.f - fy) + r1 * fy;
}

// ---------------------------------------------------------------------------
// Weight pack: fp32 (Cout, Cin*9) -> f16 fragment-ordered [Cout][Kpad] where
// each 32-K block is stored [half][16] with contiguous K per lane half
//   K = h*16 + i
// so the GEMM's A fragment is ONE contiguous 32B v16h load.
// ---------------------------------------------------------------------------
__global__ void ail_pack_w(const float* __restrict__ w, _Float16* __restrict__ apk,
                           int Cout, int K, int Kpad)
{
    int i = blockIdx.x * blockDim.x + threadIdx.x;
    if (i >= Cout * Kpad) return;
    int m = i / Kpad, r = i % Kpad;
    int kb = r & ~31, within = r & 31;
    int h = within >> 4, ii = within & 15;
    int k = kb + h * 16 + ii;
    float v = (k < K) ? w[(size_t)m * K + k] : 0.f;
    apk[i] = (_Float16)v;
}

// ---------------------------------------------------------------------------
// im2col (3x3, stride 2, pad 1) to f16, K zero-padded to Kpad (mult of 32):
//   bpk[b][pix][k]  (K contiguous -> B fragment K = half*16+i is one 32B load)
// ---------------------------------------------------------------------------
__global__ void ail_im2col(const float* __restrict__ in, _Float16* __restrict__ bpk,
                           int Bn, int Cin, int H, int W, int OH, int OW,
                           int K, int Kpad)
{
    int i = blockIdx.x * blockDim.x + threadIdx.x;
    int Npix = OH * OW;
    if (i >= Bn * Npix * Kpad) return;
    int k  = i % Kpad;
    int bp = i / Kpad;
    int pix = bp % Npix;
    int b   = bp / Npix;
    float v = 0.f;
    if (k < K) {
        int c  = k / 9;
        int rr = (k % 9) / 3;
        int ss = k % 3;
        int oh = pix / OW, ow = pix % OW;
        int ih = oh * 2 - 1 + rr;
        int iw = ow * 2 - 1 + ss;
        if ((unsigned)ih < (unsigned)H && (unsigned)iw < (unsigned)W)
            v = in[(((size_t)b * Cin + c) * H + ih) * W + iw];
    }
    bpk[i] = (_Float16)v;
}

// ---------------------------------------------------------------------------
// Packed GEMM on WMMA f16->f32: each wave owns a 16(Cout) x 64(pix) tile;
// per K-step: 1 A load + 4 B loads (contiguous 32B each) + 4 WMMAs.
// Software-pipelined one K-step deep so loads for kb+32 are in flight while
// the 4 WMMAs for kb execute (avoids s_wait_loadcnt 0 in front of every WMMA).
// Epilogue fuses bias + leaky-relu(0.2). No guards: M%16==0, N%64==0, K%32==0.
// ---------------------------------------------------------------------------
__global__ __launch_bounds__(32) void ail_gemm_wmma(
    const _Float16* __restrict__ Apk, const _Float16* __restrict__ Bpk,
    const float* __restrict__ bias, float* __restrict__ out,
    int Cout, int Npix, int Kpad)
{
    const int lane = threadIdx.x;
    const int half = lane >> 4;
    const int l    = lane & 15;
    const int mTiles = Cout >> 4;
    const int nTiles = Npix >> 6;

    int id = blockIdx.x;
    const int b  = id / (mTiles * nTiles);
    id %= (mTiles * nTiles);
    const int mt = id / nTiles;
    const int nt = id % nTiles;

    const _Float16* __restrict__ arow =
        Apk + (size_t)(mt * 16 + l) * Kpad + (half << 4);
    const _Float16* __restrict__ bcol =
        Bpk + ((size_t)b * Npix + nt * 64 + l) * Kpad + (half << 4);
    const size_t cs = (size_t)16 * Kpad;    // column-subtile stride

    v8f acc0 = {}, acc1 = {}, acc2 = {}, acc3 = {};

    // prologue: fragments for kb = 0
    v16h a  = *(const v16h*)(arow);
    v16h f0 = *(const v16h*)(bcol);
    v16h f1 = *(const v16h*)(bcol + cs);
    v16h f2 = *(const v16h*)(bcol + 2 * cs);
    v16h f3 = *(const v16h*)(bcol + 3 * cs);

    for (int kb = 32; kb < Kpad; kb += 32) {
        // issue next K-step's loads into fresh registers (in flight during WMMAs)
        v16h an = *(const v16h*)(arow + kb);
        v16h g0 = *(const v16h*)(bcol + kb);
        v16h g1 = *(const v16h*)(bcol + cs + kb);
        v16h g2 = *(const v16h*)(bcol + 2 * cs + kb);
        v16h g3 = *(const v16h*)(bcol + 3 * cs + kb);

        acc0 = __builtin_amdgcn_wmma_f32_16x16x32_f16(false, a, false, f0, (short)0, acc0, false, false);
        acc1 = __builtin_amdgcn_wmma_f32_16x16x32_f16(false, a, false, f1, (short)0, acc1, false, false);
        acc2 = __builtin_amdgcn_wmma_f32_16x16x32_f16(false, a, false, f2, (short)0, acc2, false, false);
        acc3 = __builtin_amdgcn_wmma_f32_16x16x32_f16(false, a, false, f3, (short)0, acc3, false, false);

        a = an; f0 = g0; f1 = g1; f2 = g2; f3 = g3;
    }
    // drain
    acc0 = __builtin_amdgcn_wmma_f32_16x16x32_f16(false, a, false, f0, (short)0, acc0, false, false);
    acc1 = __builtin_amdgcn_wmma_f32_16x16x32_f16(false, a, false, f1, (short)0, acc1, false, false);
    acc2 = __builtin_amdgcn_wmma_f32_16x16x32_f16(false, a, false, f2, (short)0, acc2, false, false);
    acc3 = __builtin_amdgcn_wmma_f32_16x16x32_f16(false, a, false, f3, (short)0, acc3, false, false);

#pragma unroll
    for (int r = 0; r < 8; ++r) {
        int M = mt * 16 + (half << 3) + r;          // D row layout per ISA
        float bs = bias[M];
        float* orow = out + ((size_t)b * Cout + M) * Npix + nt * 64 + l;
        float y;
        y = acc0[r] + bs; orow[0]  = (y >= 0.f) ? y : 0.2f * y;
        y = acc1[r] + bs; orow[16] = (y >= 0.f) ? y : 0.2f * y;
        y = acc2[r] + bs; orow[32] = (y >= 0.f) ? y : 0.2f * y;
        y = acc3[r] + bs; orow[48] = (y >= 0.f) ? y : 0.2f * y;
    }
}

// ---------------------------------------------------------------------------
// Instance norm: pass 1 stats (mean, rsqrt(var+eps)) per (b,c) plane.
// ---------------------------------------------------------------------------
__global__ __launch_bounds__(256) void ail_inorm_stats(
    const float* __restrict__ x, float* __restrict__ stats, int Np)
{
    const float* p = x + (size_t)blockIdx.x * Np;
    float s = 0.f, s2 = 0.f;
    for (int i = threadIdx.x; i < Np; i += 256) {
        float v = p[i]; s += v; s2 += v * v;
    }
    __shared__ float sh[256], sh2[256];
    sh[threadIdx.x] = s; sh2[threadIdx.x] = s2;
    __syncthreads();
    for (int st = 128; st > 0; st >>= 1) {
        if (threadIdx.x < st) {
            sh[threadIdx.x]  += sh[threadIdx.x + st];
            sh2[threadIdx.x] += sh2[threadIdx.x + st];
        }
        __syncthreads();
    }
    if (threadIdx.x == 0) {
        float mean = sh[0] / (float)Np;
        float var  = sh2[0] / (float)Np - mean * mean;
        stats[blockIdx.x * 2 + 0] = mean;
        stats[blockIdx.x * 2 + 1] = rsqrtf(var + 1e-5f);
    }
}

// Pass 2: in-place normalize + affine.
__global__ void ail_inorm_apply(float* __restrict__ x,
                                const float* __restrict__ stats,
                                const float* __restrict__ g,
                                const float* __restrict__ be,
                                int C, int Np, int total)
{
    int i = blockIdx.x * blockDim.x + threadIdx.x;
    if (i >= total) return;
    int bc = i / Np;
    int c  = bc % C;
    float mean = stats[bc * 2 + 0];
    float inv  = stats[bc * 2 + 1];
    x[i] = (x[i] - mean) * inv * g[c] + be[c];
}

// ---------------------------------------------------------------------------
// Head: (B,128,8,8) -> pooled x(512) -> wgen(3) & ada(96)->softmax->cumsum.
// One block of 128 threads per batch element.
// ---------------------------------------------------------------------------
__global__ __launch_bounds__(128) void ail_head(
    const float* __restrict__ act5,
    const float* __restrict__ wgen_w, const float* __restrict__ wgen_b,
    const float* __restrict__ ada_w,  const float* __restrict__ ada_b,
    float* __restrict__ wts,          // (B,3)
    float* __restrict__ verts)        // (B,3,33)
{
    __shared__ float xs[512];
    __shared__ float ada[96];
    const int b = blockIdx.x;
    const int c = threadIdx.x;        // 0..127 = channel
    const float* p = act5 + ((size_t)b * 128 + c) * 64;
#pragma unroll
    for (int pr = 0; pr < 4; ++pr) {  // (p,r) quadrant
        int pp = pr >> 1, rr = pr & 1;
        float s = 0.f;
        for (int q = 0; q < 4; ++q)
            for (int ss = 0; ss < 4; ++ss)
                s += p[(pp * 4 + q) * 8 + (rr * 4 + ss)];
        xs[c * 4 + pp * 2 + rr] = s * (1.0f / 16.0f);
    }
    __syncthreads();
    if (c < 96) {
        float s = ada_b[c];
        for (int k = 0; k < 512; ++k) s += xs[k] * ada_w[k * 96 + c];
        ada[c] = s;
    } else if (c < 99) {
        int j = c - 96;
        float s = wgen_b[j];
        for (int k = 0; k < 512; ++k) s += xs[k] * wgen_w[k * 3 + j];
        wts[b * 3 + j] = s;
    }
    __syncthreads();
    if (c < 3) {                      // softmax over 32 + cumsum -> vertices
        float mx = -1e30f;
        for (int t = 0; t < 32; ++t) mx = fmaxf(mx, ada[c * 32 + t]);
        float e[32], sum = 0.f;
        for (int t = 0; t < 32; ++t) { e[t] = expf(ada[c * 32 + t] - mx); sum += e[t]; }
        float inv = 1.0f / sum, cum = 0.f;
        float* vp = verts + ((size_t)b * 3 + c) * VV;
        vp[0] = 0.f;
        for (int t = 0; t < 32; ++t) { cum += e[t] * inv; vp[t + 1] = cum; }
    }
}

// ---------------------------------------------------------------------------
// LUT generation, channel-interleaved + padded to float4 for 1-load corners:
//   lut4[b][flat][c] = sum_k wts[b][k] * basis[k][c*V^3 + flat]
// ---------------------------------------------------------------------------
__global__ void ail_lut_gen(const float* __restrict__ wts,
                            const float* __restrict__ basis,
                            float* __restrict__ lut4, int B)
{
    int i = blockIdx.x * blockDim.x + threadIdx.x;
    int total = B * VV3;
    if (i >= total) return;
    int flat = i % VV3, b = i / VV3;
    float w0 = wts[b * 3 + 0], w1 = wts[b * 3 + 1], w2 = wts[b * 3 + 2];
    float4 o;
    float* oc = (float*)&o;
#pragma unroll
    for (int c = 0; c < 3; ++c) {
        size_t col = (size_t)c * VV3 + flat;
        oc[c] = w0 * basis[col] + w1 * basis[107811 + col] + w2 * basis[2 * 107811 + col];
    }
    o.w = 0.f;
    ((float4*)lut4)[i] = o;
}

// ---------------------------------------------------------------------------
// Final transform: streaming float4 per plane, branchless binary search on
// LDS vertices, 8 trilinear corners each a single float4 gather (L2-resident).
// ---------------------------------------------------------------------------
__device__ __forceinline__ void ail_locate(const float* __restrict__ v,
                                           float x, int& idx, float& f)
{
    x = fminf(fmaxf(x, 0.f), 1.f);
    int i = 0;                                  // largest j<=31 with v[j] <= x
    if (x >= v[i + 16]) i += 16;
    if (x >= v[i + 8])  i += 8;
    if (x >= v[i + 4])  i += 4;
    if (x >= v[i + 2])  i += 2;
    if (x >= v[i + 1])  i += 1;
    float v0 = v[i], v1 = v[i + 1];
    f = fminf(fmaxf((x - v0) / fmaxf(v1 - v0, 1e-10f), 0.f), 1.f);
    idx = i;
}

__global__ __launch_bounds__(256) void ail_apply(
    const float* __restrict__ imgs, const float* __restrict__ lut4,
    const float* __restrict__ verts, float* __restrict__ out,
    int HW, int blocksPerB)
{
    __shared__ float vs[3][VV];
    const int b  = blockIdx.x / blocksPerB;
    const int qb = blockIdx.x % blocksPerB;
    if (threadIdx.x < 99)
        ((float*)vs)[threadIdx.x] = verts[(size_t)b * 99 + threadIdx.x];
    __syncthreads();

    const int q = qb * 256 + threadIdx.x;       // quad (4 px) within batch
    const size_t base = (size_t)b * 3 * HW + (size_t)q * 4;
    const float4 r4  = *(const float4*)(imgs + base);
    const float4 g4  = *(const float4*)(imgs + base + (size_t)HW);
    const float4 bl4 = *(const float4*)(imgs + base + 2 * (size_t)HW);
    const float* rr = (const float*)&r4;
    const float* gg = (const float*)&g4;
    const float* bb = (const float*)&bl4;
    const float* lutb = lut4 + (size_t)b * VV3 * 4;

    float4 o0, o1, o2;
    float* p0 = (float*)&o0; float* p1 = (float*)&o1; float* p2 = (float*)&o2;

#pragma unroll
    for (int px = 0; px < 4; ++px) {
        int ir, ig, ib; float fr, fg, fb;
        ail_locate(vs[0], rr[px], ir, fr);
        ail_locate(vs[1], gg[px], ig, fg);
        ail_locate(vs[2], bb[px], ib, fb);
        float a0 = 0.f, a1 = 0.f, a2 = 0.f;
#pragma unroll
        for (int db = 0; db < 2; ++db) {
            float wb = db ? fb : 1.f - fb;
            int tb = (ib + db) * VV;
#pragma unroll
            for (int dg = 0; dg < 2; ++dg) {
                float wbg = wb * (dg ? fg : 1.f - fg);
                int tg = (tb + ig + dg) * VV;
#pragma unroll
                for (int dr = 0; dr < 2; ++dr) {
                    float w = wbg * (dr ? fr : 1.f - fr);
                    const float4 lv = *(const float4*)(lutb + (size_t)(tg + ir + dr) * 4);
                    a0 += w * lv.x; a1 += w * lv.y; a2 += w * lv.z;
                }
            }
        }
        p0[px] = a0; p1[px] = a1; p2[px] = a2;
    }
    *(float4*)(out + base)                    = o0;
    *(float4*)(out + base + (size_t)HW)       = o1;
    *(float4*)(out + base + 2 * (size_t)HW)   = o2;
}

// ---------------------------------------------------------------------------
// Launcher
// ---------------------------------------------------------------------------
extern "C" void kernel_launch(void* const* d_in, const int* in_sizes, int n_in,
                              void* d_out, int out_size, void* d_ws, size_t ws_size,
                              hipStream_t stream) {
    (void)in_sizes; (void)n_in; (void)out_size; (void)ws_size;
    const float* imgs   = (const float*)d_in[0];
    const float* w1 = (const float*)d_in[1];  const float* b1 = (const float*)d_in[2];
    const float* g1 = (const float*)d_in[3];  const float* be1 = (const float*)d_in[4];
    const float* w2 = (const float*)d_in[5];  const float* b2 = (const float*)d_in[6];
    const float* g2 = (const float*)d_in[7];  const float* be2 = (const float*)d_in[8];
    const float* w3 = (const float*)d_in[9];  const float* b3 = (const float*)d_in[10];
    const float* g3 = (const float*)d_in[11]; const float* be3 = (const float*)d_in[12];
    const float* w4 = (const float*)d_in[13]; const float* b4 = (const float*)d_in[14];
    const float* g4 = (const float*)d_in[15]; const float* be4 = (const float*)d_in[16];
    const float* w5 = (const float*)d_in[17]; const float* b5 = (const float*)d_in[18];
    const float* wgen_w = (const float*)d_in[19];
    const float* wgen_b = (const float*)d_in[20];
    const float* basis_w = (const float*)d_in[21];
    const float* ada_w  = (const float*)d_in[22];
    const float* ada_b  = (const float*)d_in[23];
    float* out = (float*)d_out;

    // ---- workspace carve (bytes, 256B-aligned; ~10 MB total)
    char* wsb = (char*)d_ws;
    size_t off = 0;
    auto carveB = [&](size_t bytes) {
        off = (off + 255) & ~(size_t)255;
        void* p = wsb + off; off += bytes; return p;
    };
    const int B = 2, H = 2048, W = 2048, HW = H * W;
    float* resized = (float*)carveB(sizeof(float) * 2u * 3 * 256 * 256);
    float* act1    = (float*)carveB(sizeof(float) * 2u * 16 * 128 * 128);
    float* act2    = (float*)carveB(sizeof(float) * 2u * 32 * 64 * 64);
    float* act3    = (float*)carveB(sizeof(float) * 2u * 64 * 32 * 32);
    float* act4    = (float*)carveB(sizeof(float) * 2u * 128 * 16 * 16);
    float* act5    = (float*)carveB(sizeof(float) * 2u * 128 * 8 * 8);
    float* stats   = (float*)carveB(sizeof(float) * 2u * 128 * 2);
    float* wts     = (float*)carveB(sizeof(float) * 8);
    float* verts   = (float*)carveB(sizeof(float) * 2u * 3 * VV);
    float* lut4    = (float*)carveB(sizeof(float) * 2u * VV3 * 4);
    // packed f16 weights per layer (Kpad: 32,160,288,576,1152)
    _Float16* a1p = (_Float16*)carveB(sizeof(_Float16) * 16u  * 32);
    _Float16* a2p = (_Float16*)carveB(sizeof(_Float16) * 32u  * 160);
    _Float16* a3p = (_Float16*)carveB(sizeof(_Float16) * 64u  * 288);
    _Float16* a4p = (_Float16*)carveB(sizeof(_Float16) * 128u * 576);
    _Float16* a5p = (_Float16*)carveB(sizeof(_Float16) * 128u * 1152);
    // shared im2col buffer sized for the largest layer (L2: 2*4096*160)
    _Float16* bpk = (_Float16*)carveB(sizeof(_Float16) * 2u * 4096 * 160);

    // 1) resize to 256x256
    {
        int total = B * 3 * 256 * 256;
        ail_resize<<<(total + 255) / 256, 256, 0, stream>>>(imgs, resized, B * 3, H, W, 256, 256);
    }

    // 2) pack all conv weights to fragment-ordered f16 (cheap, every call)
    auto packw = [&](const float* w, _Float16* apk, int Cout, int K, int Kpad) {
        int total = Cout * Kpad;
        ail_pack_w<<<(total + 255) / 256, 256, 0, stream>>>(w, apk, Cout, K, Kpad);
    };
    packw(w1, a1p, 16,  27,   32);
    packw(w2, a2p, 32,  144,  160);
    packw(w3, a3p, 64,  288,  288);
    packw(w4, a4p, 128, 576,  576);
    packw(w5, a5p, 128, 1152, 1152);

    // 3) conv backbone: im2col -> packed WMMA GEMM (+bias+lrelu) -> inorm
    auto conv = [&](const float* in, const _Float16* apk, const float* bias, float* o,
                    int Cin, int Hin, int Cout, int Kpad) {
        int OH = Hin / 2, Npix = OH * OH, K = Cin * 9;
        int ctotal = B * Npix * Kpad;
        ail_im2col<<<(ctotal + 255) / 256, 256, 0, stream>>>(in, bpk, B, Cin, Hin, Hin, OH, OH, K, Kpad);
        int blocks = B * (Cout / 16) * (Npix / 64);
        ail_gemm_wmma<<<blocks, 32, 0, stream>>>(apk, bpk, bias, o, Cout, Npix, Kpad);
    };
    auto inorm = [&](float* x, const float* g, const float* be, int C, int Hc) {
        int Np = Hc * Hc, total = B * C * Np;
        ail_inorm_stats<<<B * C, 256, 0, stream>>>(x, stats, Np);
        ail_inorm_apply<<<(total + 255) / 256, 256, 0, stream>>>(x, stats, g, be, C, Np, total);
    };
    conv(resized, a1p, b1, act1, 3, 256, 16, 32);      inorm(act1, g1, be1, 16, 128);
    conv(act1,    a2p, b2, act2, 16, 128, 32, 160);    inorm(act2, g2, be2, 32, 64);
    conv(act2,    a3p, b3, act3, 32, 64, 64, 288);     inorm(act3, g3, be3, 64, 32);
    conv(act3,    a4p, b4, act4, 64, 32, 128, 576);    inorm(act4, g4, be4, 128, 16);
    conv(act4,    a5p, b5, act5, 128, 16, 128, 1152);  // lrelu only, no inorm

    // 4) head: pool -> FCs -> softmax/cumsum vertices
    ail_head<<<B, 128, 0, stream>>>(act5, wgen_w, wgen_b, ada_w, ada_b, wts, verts);

    // 5) LUT generation (channel-interleaved float4)
    {
        int total = B * VV3;
        ail_lut_gen<<<(total + 255) / 256, 256, 0, stream>>>(wts, basis_w, lut4, B);
    }

    // 6) full-resolution AiLUT transform (bandwidth-bound)
    {
        int blocksPerB = (HW / 4) / 256;  // 4096
        ail_apply<<<B * blocksPerB, 256, 0, stream>>>(imgs, lut4, verts, out, HW, blocksPerB);
    }
}
